// CompressiveMemory_2164663517233
// MI455X (gfx1250) — matrix operand order repaired and checked by reading, not compiled, not run on hardware
//
#include <hip/hip_runtime.h>
#include <math.h>

// Problem constants (from the reference)
#define B_   8
#define S_   4096
#define D_   1024
#define H_   8
#define DK_  64
#define DV_  64
#define L_   256
#define NSEG (S_ / L_)     // 16
#define NBH  (B_ * H_)     // 64
#define HD_  (H_ * DK_)    // 512

typedef __attribute__((ext_vector_type(16))) _Float16 v16h;
typedef __attribute__((ext_vector_type(8)))  _Float16 v8h;
typedef __attribute__((ext_vector_type(4)))  _Float16 v4h;
typedef __attribute__((ext_vector_type(8)))  float    v8f;

static __device__ __forceinline__ float elu1f(float x) {
  return x > 0.f ? x + 1.f : __expf(x);     // elu(x)+1
}

// ---------------------------------------------------------------------------
// CDNA5 async global->LDS copy (GLOBAL_LOAD_ASYNC_TO_LDS_B128, ASYNCcnt).
// VDST VGPR carries the LDS byte offset; VADDR the 64-bit global address.
// ---------------------------------------------------------------------------
typedef __attribute__((address_space(3))) void lds_void;

static __device__ __forceinline__ void async_copy_b128(void* lds_dst, const void* g_src) {
  const unsigned loff = (unsigned)(size_t)(lds_void*)lds_dst;  // LDS byte address
  asm volatile("global_load_async_to_lds_b128 %0, %1, off"
               :: "v"(loff), "v"(g_src) : "memory");
}
static __device__ __forceinline__ void wait_async0() {
  asm volatile("s_wait_asynccnt 0x0" ::: "memory");
}
static __device__ __forceinline__ void wait_ds0() {
  asm volatile("s_wait_dscnt 0x0" ::: "memory");
}

// ---------------------------------------------------------------------------
// WMMA fragment helpers (CDNA5 16x16x32 f16, wave32 layouts per ISA 7.12.2)
// ---------------------------------------------------------------------------
// A fragment 16x32: lanes 0-15 -> row=lane, K={0..7,16..23}; lanes 16-31 ->
// row=lane-16, K={8..15,24..31}
static __device__ __forceinline__ v16h load_a_frag(const _Float16* base, int stride, int lane) {
  const int r  = lane & 15;
  const int kh = (lane >> 4) << 3;
  const _Float16* p = base + r * stride + kh;
  v16h a;
#pragma unroll
  for (int i = 0; i < 8; ++i) a[i] = p[i];
#pragma unroll
  for (int i = 0; i < 8; ++i) a[8 + i] = p[16 + i];
  return a;
}

// Same, applying elu(x)+1 while building the fragment (sq = elu(q)+1)
static __device__ __forceinline__ v16h load_a_frag_elu(const _Float16* base, int stride, int lane) {
  const int r  = lane & 15;
  const int kh = (lane >> 4) << 3;
  const _Float16* p = base + r * stride + kh;
  v16h a;
#pragma unroll
  for (int i = 0; i < 8; ++i) { float x = (float)p[i];      a[i]     = (_Float16)elu1f(x); }
#pragma unroll
  for (int i = 0; i < 8; ++i) { float x = (float)p[16 + i]; a[8 + i] = (_Float16)elu1f(x); }
  return a;
}

// B fragment 32x16 from [N][K]-major storage (K={0..7,16..23} / {8..15,24..31} per lane half, matching A)
static __device__ __forceinline__ v16h load_b_frag(const _Float16* baseT, int stride, int lane) {
  const int n  = lane & 15;
  const int kh = (lane >> 4) << 3;
  const _Float16* p = baseT + n * stride + kh;
  v16h b;
#pragma unroll
  for (int i = 0; i < 8; ++i) b[i] = p[i];
#pragma unroll
  for (int i = 0; i < 8; ++i) b[8 + i] = p[16 + i];
  return b;
}

static __device__ __forceinline__ v8f wmma32(v16h a, v16h b, v8f c) {
  return __builtin_amdgcn_wmma_f32_16x16x32_f16(false, a, false, b, (short)0, c, false, false);
}

// ---------------------------------------------------------------------------
// One-time conversion kernels (bandwidth passes, ~10us total at 23.3 TB/s)
// ---------------------------------------------------------------------------
__global__ __launch_bounds__(256)
void cvt_f32_f16_kernel(const float* __restrict__ src, _Float16* __restrict__ dst, int n) {
  const int i = ((int)blockIdx.x * 256 + (int)threadIdx.x) * 4;
  if (i + 3 < n) {
    float4 v = *(const float4*)(src + i);
    v4h o = {(_Float16)v.x, (_Float16)v.y, (_Float16)v.z, (_Float16)v.w};
    *(v4h*)(dst + i) = o;
  }
}

// WT[n][k] = (f16) W[k][n]  (pre-transpose weights so GEMM B-tiles are copies)
__global__ __launch_bounds__(256)
void transpose_cvt_kernel(const float* __restrict__ W, _Float16* __restrict__ WT, int K, int N) {
  const int idx = (int)blockIdx.x * 256 + (int)threadIdx.x;
  if (idx < K * N) {
    const int k = idx / N, n = idx % N;
    WT[(size_t)n * K + k] = (_Float16)W[idx];
  }
}

// ---------------------------------------------------------------------------
// f16 WMMA GEMM: C(MxN) = A(MxK) @ BT(NxK)^T, f32 accumulate.
// 128x128 block tile, 8 wave32 waves (2x4), each wave -> 64x32.
// Double-buffered async global->LDS staging (ASYNCcnt pipeline).
// ---------------------------------------------------------------------------
template <bool F16OUT>
__global__ __launch_bounds__(256)
void wmma_gemm_f16(const _Float16* __restrict__ A, const _Float16* __restrict__ BT,
                   float* __restrict__ Cf, _Float16* __restrict__ Ch,
                   int M, int N, int K) {
  (void)M;
  __shared__ _Float16 As[2][128][32];
  __shared__ _Float16 Bs[2][128][32];   // [n][k]

  const int tid  = (int)threadIdx.x;
  const int lane = tid & 31;
  const int wave = tid >> 5;
  const int gN = N >> 7;
  const int m0 = ((int)blockIdx.x / gN) << 7;
  const int n0 = ((int)blockIdx.x % gN) << 7;
  const int wm = (wave >> 2) << 6;
  const int wn = (wave & 3) << 5;

  // Each thread owns 16 contiguous f16 (two b128 chunks) of A and of B.
  const int srow = tid >> 1;            // 0..127
  const int scol = (tid & 1) << 4;      // 0 or 16 (f16 elements)

  auto issue_tile = [&](int buf, int kb) {
    const _Float16* ga = A  + (size_t)(m0 + srow) * K + kb + scol;
    const _Float16* gb = BT + (size_t)(n0 + srow) * K + kb + scol;
    async_copy_b128(&As[buf][srow][scol],     ga);
    async_copy_b128(&As[buf][srow][scol + 8], ga + 8);
    async_copy_b128(&Bs[buf][srow][scol],     gb);
    async_copy_b128(&Bs[buf][srow][scol + 8], gb + 8);
  };

  v8f acc[4][2];
#pragma unroll
  for (int i = 0; i < 4; ++i)
#pragma unroll
    for (int j = 0; j < 2; ++j) acc[i][j] = {};

  issue_tile(0, 0);
  int cur = 0;
  for (int kb = 0; kb < K; kb += 32) {
    wait_async0();        // this wave's async copies for buf `cur` done
    __syncthreads();      // every wave's copies done; prior reads of 1-cur done
    if (kb + 32 < K) issue_tile(1 - cur, kb + 32);

    v16h bf[2];
#pragma unroll
    for (int j = 0; j < 2; ++j) bf[j] = load_b_frag(&Bs[cur][wn + j * 16][0], 32, lane);
#pragma unroll
    for (int i = 0; i < 4; ++i) {
      v16h af = load_a_frag(&As[cur][wm + i * 16][0], 32, lane);
#pragma unroll
      for (int j = 0; j < 2; ++j) acc[i][j] = wmma32(af, bf[j], acc[i][j]);
    }
    cur ^= 1;
  }

  const int cn = lane & 15;
  const int h8 = (lane >> 4) << 3;
#pragma unroll
  for (int i = 0; i < 4; ++i)
#pragma unroll
    for (int j = 0; j < 2; ++j) {
      const size_t base = (size_t)(m0 + wm + i * 16 + h8) * N + n0 + wn + j * 16 + cn;
      if constexpr (F16OUT) {
#pragma unroll
        for (int r = 0; r < 8; ++r) Ch[base + (size_t)r * N] = (_Float16)acc[i][j][r];
      } else {
#pragma unroll
        for (int r = 0; r < 8; ++r) Cf[base + (size_t)r * N] = acc[i][j][r];
      }
    }
}

// ---------------------------------------------------------------------------
// Sequential memory scan per (b,h): store carry (mem transposed f16, z f32)
// per segment, then mem += (elu(k)+1)^T @ v ; z += sum_l(elu(k)+1).
// Tiny FLOPs (~1% of total) -> VALU.
// ---------------------------------------------------------------------------
__global__ __launch_bounds__(256)
void mem_scan_kernel(const _Float16* __restrict__ kh, const _Float16* __restrict__ vh,
                     _Float16* __restrict__ memstatesT, float* __restrict__ zstates) {
  __shared__ _Float16 sks[L_][DK_];
  __shared__ _Float16 vsh[L_][DV_];
  __shared__ float mem_s[DK_ * DV_];
  __shared__ float z_s[DK_];

  const int tid = (int)threadIdx.x;
  const int bh = (int)blockIdx.x;
  const int b = bh / H_, h = bh % H_;

  for (int i = tid; i < DK_ * DV_; i += 256) mem_s[i] = 0.f;
  if (tid < DK_) z_s[tid] = 1.f / (float)DK_;
  __syncthreads();

  for (int seg = 0; seg < NSEG; ++seg) {
    // carry state BEFORE this segment's update; mem stored transposed [dv][kk]
    _Float16* mst = memstatesT + ((size_t)seg * NBH + bh) * (DK_ * DV_);
    for (int i = tid; i < DK_ * DV_; i += 256) {
      const int kk = i >> 6, dv = i & 63;
      mst[dv * DK_ + kk] = (_Float16)mem_s[i];
    }
    if (tid < DK_) zstates[((size_t)seg * NBH + bh) * DK_ + tid] = z_s[tid];

    // stage sk = elu(k)+1 and v (one row per thread)
    {
      const int l = tid;
      const size_t row = (size_t)(b * S_ + seg * L_ + l);
      const v8h* kp = (const v8h*)(kh + row * HD_ + h * DK_);
      const v8h* vp = (const v8h*)(vh + row * HD_ + h * DV_);
#pragma unroll
      for (int i = 0; i < 8; ++i) {
        v8h kv = kp[i], vv = vp[i];
#pragma unroll
        for (int j = 0; j < 8; ++j) {
          sks[l][i * 8 + j] = (_Float16)elu1f((float)kv[j]);
          vsh[l][i * 8 + j] = vv[j];
        }
      }
    }
    __syncthreads();

#pragma unroll
    for (int i = 0; i < 16; ++i) {
      const int idx = tid + (i << 8);
      const int kk = idx >> 6, dv = idx & 63;
      float a = 0.f;
      for (int l = 0; l < L_; ++l) a += (float)sks[l][kk] * (float)vsh[l][dv];
      mem_s[idx] += a;
    }
    if (tid < DK_) {
      float a = 0.f;
      for (int l = 0; l < L_; ++l) a += (float)sks[l][tid];
      z_s[tid] += a;
    }
    __syncthreads();
  }
}

// ---------------------------------------------------------------------------
// Per-(b,h,seg) attention: WMMA QK^T -> exp -> WMMA PV, WMMA sq@mem, gated
// combine. q/k/mem tiles staged with async global->LDS copies; v transposed
// manually (no transposing async op). Output written f16 for the Wo GEMM.
// ---------------------------------------------------------------------------
__global__ __launch_bounds__(256)
void attention_kernel(const _Float16* __restrict__ qh, const _Float16* __restrict__ kh,
                      const _Float16* __restrict__ vh, const _Float16* __restrict__ memstatesT,
                      const float* __restrict__ zstates, const float* __restrict__ betas,
                      _Float16* __restrict__ abh) {
  __shared__ _Float16 qs[L_][DK_];       // 32 KB (async staged)
  __shared__ _Float16 ks[L_][DK_];       // 32 KB (async staged; [N=l][K=dk])
  __shared__ _Float16 vT[DV_][L_];       // 32 KB (manual transpose)
  __shared__ _Float16 memT[DV_][DK_];    //  8 KB (async staged; already [dv][kk])
  __shared__ float    zsh[DK_];
  __shared__ _Float16 pst[8][32][32];    // 16 KB per-wave P restage

  const int tid = (int)threadIdx.x, lane = tid & 31, wave = tid >> 5;
  const int seg = (int)blockIdx.x % NSEG;
  const int bh  = (int)blockIdx.x / NSEG;
  const int b = bh / H_, h = bh % H_;
  const int m0 = b * S_ + seg * L_;

  // async staging: thread t owns row t of q and k (8 b128 chunks each)
  {
    const _Float16* gq = qh + (size_t)(m0 + tid) * HD_ + h * DK_;
    const _Float16* gk = kh + (size_t)(m0 + tid) * HD_ + h * DK_;
#pragma unroll
    for (int j = 0; j < 8; ++j) {
      async_copy_b128(&qs[tid][j * 8], gq + j * 8);
      async_copy_b128(&ks[tid][j * 8], gk + j * 8);
    }
    const _Float16* gm = memstatesT + (size_t)(seg * NBH + bh) * (DK_ * DV_);
    _Float16* mflat = &memT[0][0];
    async_copy_b128(mflat + (tid * 2 + 0) * 8, gm + (tid * 2 + 0) * 8);
    async_copy_b128(mflat + (tid * 2 + 1) * 8, gm + (tid * 2 + 1) * 8);
  }
  // manual transpose staging of v
  {
    const v8h* vp = (const v8h*)(vh + (size_t)(m0 + tid) * HD_ + h * DV_);
#pragma unroll
    for (int i = 0; i < 8; ++i) {
      v8h vv = vp[i];
#pragma unroll
      for (int j = 0; j < 8; ++j) vT[i * 8 + j][tid] = vv[j];
    }
  }
  if (tid < DK_) zsh[tid] = zstates[(size_t)(seg * NBH + bh) * DK_ + tid];
  wait_async0();
  __syncthreads();

  const int r0 = wave << 5;
  const int h8 = (lane >> 4) << 3;
  const int cn = lane & 15;
  const float scale = 0.125f;            // 1/sqrt(DK)

  v16h aq[2][2];
#pragma unroll
  for (int t = 0; t < 2; ++t)
#pragma unroll
    for (int kk = 0; kk < 2; ++kk)
      aq[t][kk] = load_a_frag(&qs[r0 + t * 16][0] + kk * 32, DK_, lane);

  v8f oacc[2][4];
#pragma unroll
  for (int t = 0; t < 2; ++t)
#pragma unroll
    for (int nt = 0; nt < 4; ++nt) oacc[t][nt] = {};
  float rs[2][8];
#pragma unroll
  for (int t = 0; t < 2; ++t)
#pragma unroll
    for (int r = 0; r < 8; ++r) rs[t][r] = 0.f;

  // Column-blocked keys: 8 chunks of 32. Scores ~N(0,8) -> exp w/o max-shift
  // is safe in fp32 (softmax is shift-invariant).
  for (int jp = 0; jp < 8; ++jp) {
    const int j0 = jp << 5;
#pragma unroll
    for (int jt = 0; jt < 2; ++jt) {
      v16h bk0 = load_b_frag(&ks[j0 + jt * 16][0], DK_, lane);
      v16h bk1 = load_b_frag(&ks[j0 + jt * 16][0] + 32, DK_, lane);
#pragma unroll
      for (int t = 0; t < 2; ++t) {
        v8f sacc = {};
        sacc = wmma32(aq[t][0], bk0, sacc);
        sacc = wmma32(aq[t][1], bk1, sacc);
#pragma unroll
        for (int r = 0; r < 8; ++r) {
          float e = __expf(sacc[r] * scale);
          float v = e;
#pragma unroll
          for (int m = 1; m < 16; m <<= 1) v += __shfl_xor(v, m, 32);
          rs[t][r] += v;
          pst[wave][t * 16 + h8 + r][jt * 16 + cn] = (_Float16)e;
        }
      }
    }
    wait_ds0();   // same-wave LDS RAW: P restage -> A-fragment reload
    v16h bv[4];
#pragma unroll
    for (int nt = 0; nt < 4; ++nt) bv[nt] = load_b_frag(&vT[nt * 16][j0], L_, lane);
#pragma unroll
    for (int t = 0; t < 2; ++t) {
      v16h ap = load_a_frag(&pst[wave][t * 16][0], 32, lane);
#pragma unroll
      for (int nt = 0; nt < 4; ++nt) oacc[t][nt] = wmma32(ap, bv[nt], oacc[t][nt]);
    }
  }

  // memory-read denominator: den(row) = dot(elu(q_row)+1, z); one row per lane
  float den = 0.f;
  {
    const int row = r0 + lane;
    for (int d = 0; d < DK_; ++d) den += elu1f((float)qs[row][d]) * zsh[d];
  }

  v16h aqe[2][2];
#pragma unroll
  for (int t = 0; t < 2; ++t)
#pragma unroll
    for (int kk = 0; kk < 2; ++kk)
      aqe[t][kk] = load_a_frag_elu(&qs[r0 + t * 16][0] + kk * 32, DK_, lane);

#pragma unroll
  for (int t = 0; t < 2; ++t) {
#pragma unroll
    for (int nt = 0; nt < 4; ++nt) {
      v16h bm0 = load_b_frag(&memT[nt * 16][0], DK_, lane);
      v16h bm1 = load_b_frag(&memT[nt * 16][0] + 32, DK_, lane);
      v8f macc = {};
      macc = wmma32(aqe[t][0], bm0, macc);
      macc = wmma32(aqe[t][1], bm1, macc);

      const int n = nt * 16 + cn;
      const float gate = 1.f / (1.f + __expf(-betas[h * DV_ + n]));
      _Float16* outp = abh + (size_t)(m0 + r0 + t * 16 + h8) * (H_ * DV_) + h * DV_ + n;
#pragma unroll
      for (int r = 0; r < 8; ++r) {
        const int rl = t * 16 + h8 + r;
        const float dm = __shfl(den, rl, 32);
        const float am = macc[r] / dm;
        const float al = oacc[t][nt][r] / rs[t][r];
        outp[(size_t)r * (H_ * DV_)] = (_Float16)(gate * am + (1.f - gate) * al);
      }
    }
  }
}

// ---------------------------------------------------------------------------
// Launcher. Workspace (f16 unless noted):
//   xh 33.5M | WqT/WkT/WvT/WoT 0.5M each | qh/kh/vh/abh 16.8M each
//   memstatesT 4.2M | zstates 64K f32      -> ~215 MB total
// ---------------------------------------------------------------------------
extern "C" void kernel_launch(void* const* d_in, const int* in_sizes, int n_in,
                              void* d_out, int out_size, void* d_ws, size_t ws_size,
                              hipStream_t stream) {
  (void)in_sizes; (void)n_in; (void)out_size; (void)ws_size;
  const float* x     = (const float*)d_in[0];
  const float* Wq    = (const float*)d_in[1];
  const float* Wk    = (const float*)d_in[2];
  const float* Wv    = (const float*)d_in[3];
  const float* betas = (const float*)d_in[4];
  const float* Wo    = (const float*)d_in[5];
  float* out = (float*)d_out;

  const int M = B_ * S_;
  const size_t NX = (size_t)M * D_;          // 33,554,432
  const size_t NW = (size_t)D_ * HD_;        // 524,288
  const size_t NP = (size_t)M * HD_;         // 16,777,216

  _Float16* xh  = (_Float16*)d_ws;
  _Float16* wqt = xh  + NX;
  _Float16* wkt = wqt + NW;
  _Float16* wvt = wkt + NW;
  _Float16* wot = wvt + NW;
  _Float16* qh  = wot + NW;
  _Float16* kh  = qh  + NP;
  _Float16* vh  = kh  + NP;
  _Float16* abh = vh  + NP;
  _Float16* msT = abh + NP;
  float*    zs  = (float*)(msT + (size_t)NSEG * NBH * DK_ * DV_);

  const dim3 blk(256);

  // One-time conversion / weight pre-transpose (bandwidth passes)
  cvt_f32_f16_kernel<<<dim3((unsigned)(NX / 4 / 256)), blk, 0, stream>>>(x, xh, (int)NX);
  transpose_cvt_kernel<<<dim3((unsigned)(NW / 256)), blk, 0, stream>>>(Wq, wqt, D_, HD_);
  transpose_cvt_kernel<<<dim3((unsigned)(NW / 256)), blk, 0, stream>>>(Wk, wkt, D_, HD_);
  transpose_cvt_kernel<<<dim3((unsigned)(NW / 256)), blk, 0, stream>>>(Wv, wvt, D_, HD_);
  transpose_cvt_kernel<<<dim3((unsigned)(NW / 256)), blk, 0, stream>>>(Wo, wot, HD_, D_);

  // QKV projections: (32768 x 1024) @ (1024 x 512), f16 out
  wmma_gemm_f16<true><<<dim3((M / 128) * (HD_ / 128)), blk, 0, stream>>>(xh, wqt, nullptr, qh, M, HD_, D_);
  wmma_gemm_f16<true><<<dim3((M / 128) * (HD_ / 128)), blk, 0, stream>>>(xh, wkt, nullptr, kh, M, HD_, D_);
  wmma_gemm_f16<true><<<dim3((M / 128) * (HD_ / 128)), blk, 0, stream>>>(xh, wvt, nullptr, vh, M, HD_, D_);

  // Sequential memory scan -> per-segment carry states
  mem_scan_kernel<<<dim3(NBH), blk, 0, stream>>>(kh, vh, msT, zs);

  // Fully parallel per-(b,h,seg) gated attention
  attention_kernel<<<dim3(NBH * NSEG), blk, 0, stream>>>(qh, kh, vh, msT, zs, betas, abh);

  // Output projection: (32768 x 512) @ (512 x 1024), f32 out
  wmma_gemm_f16<false><<<dim3((M / 128) * (D_ / 128)), blk, 0, stream>>>(abh, wot, out, nullptr, M, D_, HD_);
}
